// MultiHeadLstmBlock_77489799954527
// MI455X (gfx1250) — matrix operand order repaired and checked by reading, not compiled, not run on hardware
//
#include <hip/hip_runtime.h>
#include <hip/hip_bf16.h>

// ---------------------------------------------------------------------------
// Types for CDNA5 WMMA
// ---------------------------------------------------------------------------
typedef __attribute__((ext_vector_type(16))) __bf16 v16bf;
typedef __attribute__((ext_vector_type(8)))  float  v8f;

union FragBF {
    v16bf v;
    unsigned u[8];
};

// Branchless fp32 -> bf16 (RNE, NaN-safe).
__device__ __forceinline__ unsigned short f2bf(float f) {
    unsigned u = __float_as_uint(f);
    unsigned rne = (u + 0x7fffu + ((u >> 16) & 1u)) >> 16;
    unsigned nan = (u >> 16) | 0x40u;
    bool is_nan  = (u & 0x7fffffffu) > 0x7f800000u;
    return (unsigned short)(is_nan ? nan : rne);
}
__device__ __forceinline__ unsigned pack2bf(float a, float b) {
    return (unsigned)f2bf(a) | ((unsigned)f2bf(b) << 16);
}

__device__ __forceinline__ float fast_sigmoid(float x) {
    return 1.0f / (1.0f + __expf(-x));
}
__device__ __forceinline__ float fast_tanh(float x) {
    float cx = fminf(fmaxf(x, -15.0f), 15.0f);
    float e  = __expf(-2.0f * cx);
    return (1.0f - e) / (1.0f + e);
}

// ---------------------------------------------------------------------------
// CDNA5 async global->LDS DMA (ASYNCcnt-tracked), GV addressing.
// ---------------------------------------------------------------------------
__device__ __forceinline__ void async_load_b128(void* lds_ptr, const void* gptr) {
    unsigned lds_off = (unsigned)(size_t)lds_ptr;
    asm volatile("global_load_async_to_lds_b128 %0, %1, off"
                 :: "v"(lds_off), "v"(gptr)
                 : "memory");
}
__device__ __forceinline__ void wait_async0() {
    asm volatile("s_wait_asynccnt 0x0" ::: "memory");
}

// A fragment: 16x32 bf16; rowp -> (row m = lane%16, k=0), row-major LDS.
__device__ __forceinline__ v16bf load_a(const unsigned short* rowp, int half) {
    FragBF f;
    *(uint4*)&f.u[0] = *(const uint4*)(rowp + 8 * half);
    *(uint4*)&f.u[4] = *(const uint4*)(rowp + 16 + 8 * half);
    return f.v;
}
// B fragment: 32x16 bf16 where B[k][n] = W[n][k]; rowp -> row n (= lane%16), k=0.
__device__ __forceinline__ v16bf load_b(const unsigned short* rowp, int half) {
    FragBF f;
    const unsigned short* p = rowp + 8 * half;
    *(uint4*)&f.u[0] = *(const uint4*)(p);
    *(uint4*)&f.u[4] = *(const uint4*)(p + 16);
    return f.v;
}

#define WMMA_BF16(a, b, c) \
    __builtin_amdgcn_wmma_f32_16x16x32_bf16(false, (a), false, (b), (short)0, (c), false, false)

// ---------------------------------------------------------------------------
// Problem constants
// ---------------------------------------------------------------------------
static constexpr int  Bn  = 128;
static constexpr int  T   = 197;
static constexpr int  D   = 768;
static constexpr int  H   = 768;
static constexpr int  BT  = Bn * T;            // 25216
static constexpr int  NH  = 12;
static constexpr int  HPH = 64;

static constexpr size_t PREW_E  = (size_t)D * D;
static constexpr size_t PROJW_E = (size_t)D * 2 * H;
static constexpr size_t WCOMB_E = (size_t)2 * NH * 256 * 128;
static constexpr size_t XPRE_E  = (size_t)BT * D;
static constexpr size_t LOUT_E  = (size_t)BT * 2 * H;

// ---------------------------------------------------------------------------
// Kernel 1: weight conversion / fusion
// ---------------------------------------------------------------------------
__global__ __launch_bounds__(256) void prep_kernel(
    const float* __restrict__ pre_w, const float* __restrict__ proj_w,
    const float* __restrict__ w_ih_f, const float* __restrict__ w_hh_f,
    const float* __restrict__ b_ih_f, const float* __restrict__ b_hh_f,
    const float* __restrict__ w_ih_r, const float* __restrict__ w_hh_r,
    const float* __restrict__ b_ih_r, const float* __restrict__ b_hh_r,
    unsigned short* __restrict__ preW, unsigned short* __restrict__ projW,
    unsigned short* __restrict__ Wcomb, float* __restrict__ biasS)
{
    int i = blockIdx.x * blockDim.x + threadIdx.x;

    if (i < (int)PREW_E)  preW[i]  = f2bf(pre_w[i]);
    if (i < (int)PROJW_E) projW[i] = f2bf(proj_w[i]);

    if (i < (int)WCOMB_E) {
        int k   = i & 127;
        int j   = (i >> 7) & 255;
        int blk = i >> 15;
        int d   = blk / NH;
        int hd  = blk % NH;
        int g   = j >> 6;
        int jj  = j & 63;
        int row = g * H + hd * HPH + jj;
        const float* wih = d ? w_ih_r : w_ih_f;
        const float* whh = d ? w_hh_r : w_hh_f;
        float v = (k < 64) ? wih[(size_t)row * D + hd * HPH + k]
                           : whh[(size_t)row * H + hd * HPH + (k - 64)];
        Wcomb[i] = f2bf(v);
    }
    if (i < 2 * NH * 256) {
        int j   = i & 255;
        int blk = i >> 8;
        int d   = blk / NH;
        int hd  = blk % NH;
        int g   = j >> 6;
        int jj  = j & 63;
        int row = g * H + hd * HPH + jj;
        const float* bih = d ? b_ih_r : b_ih_f;
        const float* bhh = d ? b_hh_r : b_hh_f;
        biasS[i] = bih[row] + bhh[row];
    }
}

// ---------------------------------------------------------------------------
// Kernel 2: pre-proj GEMM  Xpre[BT,768] = bf16( x @ pre_w.T + pre_b )
// Double-buffered LDS: B tile DMA'd async, A tile converted via VGPRs, both
// for chunk i+1 issued/staged while chunk i is consumed by WMMA.
// ---------------------------------------------------------------------------
__global__ __launch_bounds__(256) void pre_gemm(
    const float* __restrict__ X, const unsigned short* __restrict__ Wbf,
    const float* __restrict__ bias, unsigned short* __restrict__ Out)
{
    __shared__ unsigned short As[2][128 * 32];
    __shared__ unsigned short Bs[2][128 * 32];

    const int tid  = threadIdx.x;
    const int lane = tid & 31;
    const int wave = tid >> 5;
    const int wr   = wave >> 1, wc = wave & 1;
    const int half = (lane >> 4) & 1;
    const int ln   = lane & 15;
    const int m0   = blockIdx.x * 128;
    const int n0   = blockIdx.y * 128;
    const int NIT  = D / 32;   // 24

    auto issueB = [&](int buf, int kc) {
#pragma unroll
        for (int it = 0; it < 2; ++it) {
            int lin = tid + it * 256;
            int row = lin >> 2;
            int kq  = (lin & 3) * 8;
            async_load_b128(&Bs[buf][row * 32 + kq],
                            Wbf + (size_t)(n0 + row) * D + kc + kq);
        }
    };
    auto stageA = [&](int buf, int kc) {
#pragma unroll
        for (int it = 0; it < 4; ++it) {
            int lin = tid + it * 256;
            int row = lin >> 3;
            int kq  = (lin & 7) * 4;
            float4 v = *(const float4*)(X + (size_t)(m0 + row) * D + kc + kq);
            uint2 pk;
            pk.x = pack2bf(v.x, v.y);
            pk.y = pack2bf(v.z, v.w);
            *(uint2*)(&As[buf][row * 32 + kq]) = pk;
        }
    };

    const v8f vz = {0.f, 0.f, 0.f, 0.f, 0.f, 0.f, 0.f, 0.f};
    v8f acc[2][4];
#pragma unroll
    for (int mt = 0; mt < 2; ++mt)
#pragma unroll
        for (int jt = 0; jt < 4; ++jt) acc[mt][jt] = vz;

    issueB(0, 0);
    stageA(0, 0);
    wait_async0();
    __syncthreads();

    for (int i = 0; i < NIT; ++i) {
        const int  kc   = i * 32;
        const int  cur  = i & 1;
        const int  nxt  = cur ^ 1;
        const bool more = (i + 1 < NIT);

        if (more) issueB(nxt, kc + 32);        // DMA overlaps with WMMA below
#pragma unroll
        for (int mt = 0; mt < 2; ++mt) {
            v16bf a = load_a(&As[cur][(wr * 32 + mt * 16 + ln) * 32], half);
#pragma unroll
            for (int jt = 0; jt < 4; ++jt) {
                v16bf b = load_b(&Bs[cur][(wc * 64 + jt * 16 + ln) * 32], half);
                acc[mt][jt] = WMMA_BF16(a, b, acc[mt][jt]);
            }
        }
        if (more) {
            stageA(nxt, kc + 32);              // global loads overlap with WMMA
            wait_async0();
            __syncthreads();
        }
    }

#pragma unroll
    for (int mt = 0; mt < 2; ++mt)
#pragma unroll
        for (int jt = 0; jt < 4; ++jt) {
            int col  = n0 + wc * 64 + jt * 16 + ln;
            float bv = bias[col];
#pragma unroll
            for (int r = 0; r < 8; ++r) {
                int row = m0 + wr * 32 + mt * 16 + r + 8 * half;
                Out[(size_t)row * D + col] = f2bf(acc[mt][jt][r] + bv);
            }
        }
}

// ---------------------------------------------------------------------------
// Kernel 3: bidirectional block-diagonal LSTM recurrence.
// 192 blocks (2 dir x 12 head x 8 batch-chunks), 128 threads = 4 waves.
// Double-buffered A (=[x_t|h]); x_{t+1} DMA'd during step t's gate GEMM;
// h written to both buffers. Weights DMA'd once into LDS.
// ---------------------------------------------------------------------------
__global__ __launch_bounds__(128) void lstm_kernel(
    const unsigned short* __restrict__ Xpre,
    const unsigned short* __restrict__ Wcomb,
    const float* __restrict__ biasS,
    unsigned short* __restrict__ Lout)
{
    __shared__ unsigned short W_lds[256 * 128];    // 64 KB
    __shared__ unsigned short A_lds[2][16 * 128];  // [buf][m][ x | h ]
    __shared__ float          G_lds[16 * 256];

    const int tid  = threadIdx.x;
    const int lane = tid & 31;
    const int wave = tid >> 5;       // == gate id (i,f,g,o)
    const int half = (lane >> 4) & 1;
    const int ln   = lane & 15;

    const int bx    = blockIdx.x;
    const int dir   = bx / 96;
    const int rem   = bx % 96;
    const int hd    = rem >> 3;
    const int chunk = rem & 7;
    const int b0    = chunk * 16;

    const int erow = tid >> 3;          // 0..15
    const int ej   = (tid & 7) * 8;     // 0..56
    const int xrow = tid >> 3;
    const int xq   = (tid & 7) * 8;

    // DMA fused recurrent weights for (dir, head) into LDS
    const unsigned short* wsrc = Wcomb + (size_t)(dir * NH + hd) * 256 * 128;
    for (int i = tid; i < 256 * 128 / 8; i += 128)
        async_load_b128(W_lds + i * 8, wsrc + i * 8);

    // zero both A buffers (h region must start at 0)
    uint4 z4; z4.x = z4.y = z4.z = z4.w = 0u;
    for (int i = tid; i < 2 * 16 * 128 / 8; i += 128)
        *(uint4*)(&A_lds[0][0] + i * 8) = z4;

    // DMA x for the first step into buffer 0
    {
        const int t0 = dir ? (T - 1) : 0;
        const unsigned short* xsrc =
            Xpre + ((size_t)(b0 + xrow) * T + t0) * D + hd * HPH + xq;
        async_load_b128(&A_lds[0][xrow * 128 + xq], xsrc);
    }

    float bcol[4];
    const float* bptr = biasS + (dir * NH + hd) * 256;
#pragma unroll
    for (int jt = 0; jt < 4; ++jt) bcol[jt] = bptr[wave * 64 + jt * 16 + ln];

    float c[8];
#pragma unroll
    for (int e = 0; e < 8; ++e) c[e] = 0.f;

    wait_async0();
    __syncthreads();

    for (int s = 0; s < T; ++s) {
        const int t   = dir ? (T - 1 - s) : s;
        const int cur = s & 1;
        const int nxt = cur ^ 1;

        // prefetch x_{t+1} into the idle buffer (overlaps with gate GEMM)
        if (s + 1 < T) {
            const int tn = dir ? (T - 2 - s) : (s + 1);
            const unsigned short* xsrc =
                Xpre + ((size_t)(b0 + xrow) * T + tn) * D + hd * HPH + xq;
            async_load_b128(&A_lds[nxt][xrow * 128 + xq], xsrc);
        }

        // gates[16,256] = [x_t|h] @ Wcomb.T, C initialized with bias
        v8f acc[4];
#pragma unroll
        for (int jt = 0; jt < 4; ++jt) {
            v8f tacc;
#pragma unroll
            for (int r = 0; r < 8; ++r) tacc[r] = bcol[jt];
            acc[jt] = tacc;
        }
#pragma unroll
        for (int ks = 0; ks < 4; ++ks) {
            v16bf a = load_a(&A_lds[cur][ln * 128 + ks * 32], half);
#pragma unroll
            for (int jt = 0; jt < 4; ++jt) {
                v16bf b = load_b(W_lds + (wave * 64 + jt * 16 + ln) * 128 + ks * 32, half);
                acc[jt] = WMMA_BF16(a, b, acc[jt]);
            }
        }
        // activations -> G_lds (wave 2 = cell gate -> tanh, else sigmoid)
#pragma unroll
        for (int jt = 0; jt < 4; ++jt)
#pragma unroll
            for (int r = 0; r < 8; ++r) {
                float v = acc[jt][r];
                v = (wave == 2) ? fast_tanh(v) : fast_sigmoid(v);
                G_lds[(r + 8 * half) * 256 + wave * 64 + jt * 16 + ln] = v;
            }
        __syncthreads();

        // elementwise c/h update
        union { unsigned short s[8]; uint4 q; } hp;
#pragma unroll
        for (int e = 0; e < 8; ++e) {
            int j = ej + e;
            float gi = G_lds[erow * 256 +       j];
            float gf = G_lds[erow * 256 +  64 + j];
            float gg = G_lds[erow * 256 + 128 + j];
            float go = G_lds[erow * 256 + 192 + j];
            c[e] = gf * c[e] + gi * gg;
            hp.s[e] = f2bf(go * fast_tanh(c[e]));
        }
        *(uint4*)(&A_lds[0][erow * 128 + 64 + ej]) = hp.q;  // h into both buffers
        *(uint4*)(&A_lds[1][erow * 128 + 64 + ej]) = hp.q;
        unsigned short* gdst =
            Lout + ((size_t)(b0 + erow) * T + t) * (2 * H) + dir * H + hd * HPH + ej;
        *(uint4*)gdst = hp.q;

        wait_async0();        // x_{t+1} DMA has had the whole step to land
        __syncthreads();
    }
}

// ---------------------------------------------------------------------------
// Kernel 4: output projection  out[BT,768] = Lout[BT,1536] @ proj_w.T + proj_b
// Double-buffered; A and B tiles both async-DMA'd to LDS.
// ---------------------------------------------------------------------------
__global__ __launch_bounds__(256) void proj_gemm(
    const unsigned short* __restrict__ Abf, const unsigned short* __restrict__ Wbf,
    const float* __restrict__ bias, float* __restrict__ Out)
{
    __shared__ unsigned short As[2][128 * 32];
    __shared__ unsigned short Bs[2][128 * 32];

    const int tid  = threadIdx.x;
    const int lane = tid & 31;
    const int wave = tid >> 5;
    const int wr   = wave >> 1, wc = wave & 1;
    const int half = (lane >> 4) & 1;
    const int ln   = lane & 15;
    const int m0   = blockIdx.x * 128;
    const int n0   = blockIdx.y * 128;
    const int K    = 2 * H;     // 1536
    const int NIT  = K / 32;    // 48

    auto issueAB = [&](int buf, int kc) {
#pragma unroll
        for (int it = 0; it < 2; ++it) {
            int lin = tid + it * 256;
            int row = lin >> 2;
            int kq  = (lin & 3) * 8;
            async_load_b128(&As[buf][row * 32 + kq],
                            Abf + (size_t)(m0 + row) * K + kc + kq);
            async_load_b128(&Bs[buf][row * 32 + kq],
                            Wbf + (size_t)(n0 + row) * K + kc + kq);
        }
    };

    const v8f vz = {0.f, 0.f, 0.f, 0.f, 0.f, 0.f, 0.f, 0.f};
    v8f acc[2][4];
#pragma unroll
    for (int mt = 0; mt < 2; ++mt)
#pragma unroll
        for (int jt = 0; jt < 4; ++jt) acc[mt][jt] = vz;

    issueAB(0, 0);
    wait_async0();
    __syncthreads();

    for (int i = 0; i < NIT; ++i) {
        const int  kc   = i * 32;
        const int  cur  = i & 1;
        const int  nxt  = cur ^ 1;
        const bool more = (i + 1 < NIT);

        if (more) issueAB(nxt, kc + 32);       // DMA overlaps with WMMA below
#pragma unroll
        for (int mt = 0; mt < 2; ++mt) {
            v16bf a = load_a(&As[cur][(wr * 32 + mt * 16 + ln) * 32], half);
#pragma unroll
            for (int jt = 0; jt < 4; ++jt) {
                v16bf b = load_b(&Bs[cur][(wc * 64 + jt * 16 + ln) * 32], half);
                acc[mt][jt] = WMMA_BF16(a, b, acc[mt][jt]);
            }
        }
        if (more) {
            wait_async0();
            __syncthreads();
        }
    }

#pragma unroll
    for (int mt = 0; mt < 2; ++mt)
#pragma unroll
        for (int jt = 0; jt < 4; ++jt) {
            int col  = n0 + wc * 64 + jt * 16 + ln;
            float bv = bias[col];
#pragma unroll
            for (int r = 0; r < 8; ++r) {
                int row = m0 + wr * 32 + mt * 16 + r + 8 * half;
                Out[(size_t)row * D + col] = acc[mt][jt][r] + bv;
            }
        }
}

// ---------------------------------------------------------------------------
// Host-side launcher
// ---------------------------------------------------------------------------
extern "C" void kernel_launch(void* const* d_in, const int* in_sizes, int n_in,
                              void* d_out, int out_size, void* d_ws, size_t ws_size,
                              hipStream_t stream) {
    const float* x      = (const float*)d_in[0];
    const float* pre_w  = (const float*)d_in[1];
    const float* pre_b  = (const float*)d_in[2];
    const float* w_ih_f = (const float*)d_in[3];
    const float* w_hh_f = (const float*)d_in[4];
    const float* b_ih_f = (const float*)d_in[5];
    const float* b_hh_f = (const float*)d_in[6];
    const float* w_ih_r = (const float*)d_in[7];
    const float* w_hh_r = (const float*)d_in[8];
    const float* b_ih_r = (const float*)d_in[9];
    const float* b_hh_r = (const float*)d_in[10];
    const float* proj_w = (const float*)d_in[11];
    const float* proj_b = (const float*)d_in[12];
    float* out = (float*)d_out;

    unsigned short* preW  = (unsigned short*)d_ws;
    unsigned short* projW = preW  + PREW_E;
    unsigned short* Wcomb = projW + PROJW_E;
    unsigned short* Xpre  = Wcomb + WCOMB_E;
    unsigned short* LoutB = Xpre  + XPRE_E;
    float*          biasS = (float*)(LoutB + LOUT_E);

    {
        int total = (int)PROJW_E;
        int grid  = (total + 255) / 256;
        prep_kernel<<<grid, 256, 0, stream>>>(pre_w, proj_w,
                                              w_ih_f, w_hh_f, b_ih_f, b_hh_f,
                                              w_ih_r, w_hh_r, b_ih_r, b_hh_r,
                                              preW, projW, Wcomb, biasS);
    }
    pre_gemm<<<dim3(BT / 128, D / 128), 256, 0, stream>>>(x, preW, pre_b, Xpre);
    lstm_kernel<<<192, 128, 0, stream>>>(Xpre, Wcomb, biasS, LoutB);
    proj_gemm<<<dim3(BT / 128, D / 128), 256, 0, stream>>>(LoutB, projW, proj_b, out);
}
